// QuantizeMultiHeadAttention_57818849739284
// MI455X (gfx1250) — compile-verified, matrix operand order checked by reading, not run on hardware
//
#include <hip/hip_runtime.h>

typedef __attribute__((ext_vector_type(16))) __bf16 v16bf;
typedef __attribute__((ext_vector_type(8)))  __bf16 v8bf;
typedef __attribute__((ext_vector_type(8)))  float  v8f;
typedef int v4i __attribute__((vector_size(4 * sizeof(int))));
typedef __attribute__((address_space(1))) v4i av4i_g;   // global
typedef __attribute__((address_space(3))) v4i av4i_l;   // LDS

#define WMMA_BF16(a,b,c) __builtin_amdgcn_wmma_f32_16x16x32_bf16(false,(a),false,(b),(short)0,(c),false,false)

#if defined(__gfx1250__) && __has_builtin(__builtin_amdgcn_global_load_async_to_lds_b128)
#define USE_ASYNC_LDS 1
#else
#define USE_ASYNC_LDS 0
#endif

__device__ __forceinline__ void async_copy_b128(const __bf16* g, __bf16* l) {
#if USE_ASYNC_LDS
  __builtin_amdgcn_global_load_async_to_lds_b128(
      (av4i_g*)(void*)g, (av4i_l*)(void*)l, 0, 0);
#endif
}

__device__ __forceinline__ void wait_async() {
#if USE_ASYNC_LDS
#if __has_builtin(__builtin_amdgcn_s_wait_asynccnt)
  __builtin_amdgcn_s_wait_asynccnt(0);
#else
  asm volatile("s_wait_asynccnt 0x0" ::: "memory");
#endif
#endif
}

__device__ __forceinline__ float lsq_fwd(float x, float s) {
  // forward of LSQ: s==alpha, round-to-nearest-even of clipped x/s
  float xs = fminf(fmaxf(x / s, -128.f), 127.f);
  return rintf(xs) * s;
}

// ---------------- elementwise ----------------
__global__ void k_quant_w(const float* __restrict__ w, const float* __restrict__ alpha,
                          __bf16* __restrict__ out, int n) {
  int i = blockIdx.x * blockDim.x + threadIdx.x;
  if (i < n) out[i] = (__bf16)lsq_fwd(w[i], alpha[0]);
}

__global__ void k_cvt_bf16(const float* __restrict__ x, __bf16* __restrict__ y, int n) {
  int i = blockIdx.x * blockDim.x + threadIdx.x;
  if (i < n) y[i] = (__bf16)x[i];
}

// ---------------- generic C = A(MxK) @ B(NxK)^T, bf16 in, fp32 acc ----------------
// mode 0: outb = bf16(lsq(acc, alpha))                       (q/k projections)
// mode 1: outT[b][n][l] = bf16(lsq(acc, alpha))  transposed  (v projection -> VPT)
// mode 2: outf = lsq(acc, alpha) + resid                     (fc + residual)
// mode 3: outb = bf16(acc + bias[n])                         (fa1 / fa2)
// mode 4: outf = acc + bias[n] + resid                       (fa3 + skip)
__global__ __launch_bounds__(256) void k_gemm_xwt(
    const __bf16* __restrict__ A, const __bf16* __restrict__ Bw,
    int M, int N, int K, int mode,
    const float* __restrict__ alpha, const float* __restrict__ bias,
    const float* __restrict__ resid,
    float* __restrict__ outf, __bf16* __restrict__ outb, __bf16* __restrict__ outT)
{
  __shared__ __align__(32) __bf16 As[128 * 32];
  __shared__ __align__(32) __bf16 Bs[64 * 32];
  const int tid  = threadIdx.x;
  const int wave = tid >> 5, lane = tid & 31;
  const int nblk = N >> 6;
  const int bm = blockIdx.x / nblk, bn = blockIdx.x - bm * nblk;
  const int m0 = bm << 7, n0 = bn << 6;
  const int wm0 = (wave >> 1) << 5;   // 0,32,64,96
  const int wn0 = (wave & 1) << 5;    // 0,32
  const int fr = lane & 15, fh = (lane >> 4) << 4;
  // uniform (divergence-free) staging assignment:
  //   A tile 128x32 (8KB): thread t moves 32B: row=t>>1, col=(t&1)*16
  //   B tile  64x32 (4KB): thread t moves 16B: row=t>>2, col=(t&3)*8
  const int ra = tid >> 1, ca = (tid & 1) << 4;
  const int rb = tid >> 2, cb = (tid & 3) << 3;
  v8f acc[2][2] = {};
  for (int k0 = 0; k0 < K; k0 += 32) {
    const __bf16* ga = &A[(size_t)(m0 + ra) * K + k0 + ca];
    const __bf16* gb = &Bw[(size_t)(n0 + rb) * K + k0 + cb];
    __bf16* la = &As[ra * 32 + ca];
    __bf16* lb = &Bs[rb * 32 + cb];
#if USE_ASYNC_LDS
    async_copy_b128(ga, la);
    async_copy_b128(ga + 8, la + 8);
    async_copy_b128(gb, lb);
#else
    *(v16bf*)la = *(const v16bf*)ga;
    *(v8bf*)lb = *(const v8bf*)gb;
#endif
    if (k0 + 32 < K) {
      __builtin_prefetch(ga + 32, 0, 1);
      __builtin_prefetch(gb + 32, 0, 1);
    }
    wait_async();
    __syncthreads();
    v16bf bf0 = *(const v16bf*)&Bs[(wn0 + fr) * 32 + fh];
    v16bf bf1 = *(const v16bf*)&Bs[(wn0 + 16 + fr) * 32 + fh];
#pragma unroll
    for (int i = 0; i < 2; ++i) {
      v16bf af = *(const v16bf*)&As[(wm0 + (i << 4) + fr) * 32 + fh];
      acc[i][0] = WMMA_BF16(af, bf0, acc[i][0]);
      acc[i][1] = WMMA_BF16(af, bf1, acc[i][1]);
    }
    __syncthreads();
  }
  const int cn = lane & 15, cmh = (lane >> 4) << 3;
#pragma unroll
  for (int i = 0; i < 2; ++i)
#pragma unroll
    for (int j = 0; j < 2; ++j)
#pragma unroll
      for (int r = 0; r < 8; ++r) {
        float vv = acc[i][j][r];
        const int m = m0 + wm0 + (i << 4) + cmh + r;
        const int n = n0 + wn0 + (j << 4) + cn;
        const size_t idx = (size_t)m * N + n;
        if (mode == 0) {
          outb[idx] = (__bf16)lsq_fwd(vv, alpha[0]);
        } else if (mode == 1) {
          float qv = lsq_fwd(vv, alpha[0]);
          outT[((size_t)(m >> 10) << 20) + (size_t)n * 1024 + (m & 1023)] = (__bf16)qv;
        } else if (mode == 2) {
          outf[idx] = lsq_fwd(vv, alpha[0]) + resid[idx];
        } else if (mode == 3) {
          outb[idx] = (__bf16)(vv + bias[n]);
        } else {
          outf[idx] = vv + bias[n] + resid[idx];
        }
      }
}

// ---------------- attention pass 1: per-row online max / sum(exp) ----------------
__global__ __launch_bounds__(256) void k_attn_stats(
    const __bf16* __restrict__ QP, const __bf16* __restrict__ KP,
    const int* __restrict__ mask,
    float* __restrict__ rowmax, float* __restrict__ rowsum)
{
  const int qblk = blockIdx.x & 7;
  const int bh = blockIdx.x >> 3;
  const int b = bh >> 4, h = bh & 15;
  const int wave = threadIdx.x >> 5, lane = threadIdx.x & 31;
  const int q0 = qblk * 128 + wave * 16;
  const int fr = lane & 15, fh = (lane >> 4) << 4;
  const __bf16* Qb = QP + ((size_t)(b * 1024 + q0)) * 1024 + h * 64;
  const __bf16* Kb = KP + ((size_t)(b * 1024)) * 1024 + h * 64;
  const int* mk = mask + b * 1024;
  v16bf qf0 = *(const v16bf*)&Qb[(size_t)fr * 1024 + fh];
  v16bf qf1 = *(const v16bf*)&Qb[(size_t)fr * 1024 + 32 + fh];
  float mrow[8], lrow[8];
#pragma unroll
  for (int r = 0; r < 8; ++r) { mrow[r] = -3.0e38f; lrow[r] = 0.f; }
  for (int kb = 0; kb < 1024; kb += 16) {
    const __bf16* Kt = Kb + (size_t)kb * 1024;
    v16bf kf0 = *(const v16bf*)&Kt[(size_t)fr * 1024 + fh];
    v16bf kf1 = *(const v16bf*)&Kt[(size_t)fr * 1024 + 32 + fh];
    v8f s = {};
    s = WMMA_BF16(qf0, kf0, s);
    s = WMMA_BF16(qf1, kf1, s);
    const bool msk = (mk[kb + (lane & 15)] == 0);
#pragma unroll
    for (int r = 0; r < 8; ++r) {
      float v = msk ? -1.0e9f : s[r] * 0.125f;
      float mx = v;
      mx = fmaxf(mx, __shfl_xor(mx, 1, 32));
      mx = fmaxf(mx, __shfl_xor(mx, 2, 32));
      mx = fmaxf(mx, __shfl_xor(mx, 4, 32));
      mx = fmaxf(mx, __shfl_xor(mx, 8, 32));
      float mnew = fmaxf(mrow[r], mx);
      float p = __expf(v - mnew);
      p += __shfl_xor(p, 1, 32);
      p += __shfl_xor(p, 2, 32);
      p += __shfl_xor(p, 4, 32);
      p += __shfl_xor(p, 8, 32);
      lrow[r] = lrow[r] * __expf(mrow[r] - mnew) + p;
      mrow[r] = mnew;
    }
  }
  if ((lane & 15) == 0) {
    const int rb = bh * 1024 + q0 + ((lane >> 4) << 3);
#pragma unroll
    for (int r = 0; r < 8; ++r) { rowmax[rb + r] = mrow[r]; rowsum[rb + r] = lrow[r]; }
  }
}

// ---------------- attention pass 2: write P once + accumulate P@V ----------------
__global__ __launch_bounds__(256) void k_attn_pv(
    const __bf16* __restrict__ QP, const __bf16* __restrict__ KP,
    const __bf16* __restrict__ VPT, const int* __restrict__ mask,
    const float* __restrict__ rowmax, const float* __restrict__ rowsum,
    float* __restrict__ attn, __bf16* __restrict__ AO)
{
  __shared__ __align__(32) __bf16 Ps[8][16 * 32];
  const int qblk = blockIdx.x & 7;
  const int bh = blockIdx.x >> 3;
  const int b = bh >> 4, h = bh & 15;
  const int wave = threadIdx.x >> 5, lane = threadIdx.x & 31;
  const int q0 = qblk * 128 + wave * 16;
  const int fr = lane & 15, fh = (lane >> 4) << 4;
  const int cmh = (lane >> 4) << 3;
  const __bf16* Qb = QP + ((size_t)(b * 1024 + q0)) * 1024 + h * 64;
  const __bf16* Kb = KP + ((size_t)(b * 1024)) * 1024 + h * 64;
  const __bf16* Vb = VPT + ((size_t)b << 20) + (size_t)(h * 64) * 1024;
  const int* mk = mask + b * 1024;
  v16bf qf0 = *(const v16bf*)&Qb[(size_t)fr * 1024 + fh];
  v16bf qf1 = *(const v16bf*)&Qb[(size_t)fr * 1024 + 32 + fh];
  float mrow[8], rinv[8];
  {
    const int rb = bh * 1024 + q0 + cmh;
#pragma unroll
    for (int r = 0; r < 8; ++r) { mrow[r] = rowmax[rb + r]; rinv[r] = 1.f / rowsum[rb + r]; }
  }
  v8f oacc[4] = {};
  __bf16* myP = &Ps[wave][0];
  for (int kb = 0; kb < 1024; kb += 32) {
#pragma unroll
    for (int j = 0; j < 2; ++j) {
      const int kc = kb + (j << 4);
      const __bf16* Kt = Kb + (size_t)kc * 1024;
      v16bf kf0 = *(const v16bf*)&Kt[(size_t)fr * 1024 + fh];
      v16bf kf1 = *(const v16bf*)&Kt[(size_t)fr * 1024 + 32 + fh];
      v8f s = {};
      s = WMMA_BF16(qf0, kf0, s);
      s = WMMA_BF16(qf1, kf1, s);
      const bool msk = (mk[kc + (lane & 15)] == 0);
      const int kpos = kc + (lane & 15);
#pragma unroll
      for (int r = 0; r < 8; ++r) {
        float v = msk ? -1.0e9f : s[r] * 0.125f;
        float p = __expf(v - mrow[r]) * rinv[r];
        const int qrow = q0 + cmh + r;
        attn[((size_t)bh << 20) + (size_t)qrow * 1024 + kpos] = p;
        myP[(cmh + r) * 32 + (j << 4) + (lane & 15)] = (__bf16)p;
      }
    }
    __syncthreads();
    v16bf af = *(const v16bf*)&myP[fr * 32 + fh];
#pragma unroll
    for (int t = 0; t < 4; ++t) {
      v16bf bfv = *(const v16bf*)&Vb[(size_t)((t << 4) + fr) * 1024 + kb + fh];
      oacc[t] = WMMA_BF16(af, bfv, oacc[t]);
    }
    __syncthreads();
  }
#pragma unroll
  for (int t = 0; t < 4; ++t)
#pragma unroll
    for (int r = 0; r < 8; ++r) {
      const int qrow = q0 + cmh + r;
      const int n = h * 64 + (t << 4) + (lane & 15);
      AO[(size_t)(b * 1024 + qrow) * 1024 + n] = (__bf16)oacc[t][r];
    }
}

// ---------------- deterministic per-channel mean/var reduction ----------------
__global__ __launch_bounds__(256) void k_colstats(const float* __restrict__ y, float* __restrict__ part)
{
  const int t = threadIdx.x;
  float s0 = 0, s1 = 0, s2 = 0, s3 = 0, q0 = 0, q1 = 0, q2 = 0, q3 = 0;
  const int r0 = blockIdx.x << 8;
  for (int r = 0; r < 256; ++r) {
    const float* row = y + ((size_t)(r0 + r) << 10);
    float a = row[t], b = row[t + 256], c = row[t + 512], d = row[t + 768];
    s0 += a; s1 += b; s2 += c; s3 += d;
    q0 += a * a; q1 += b * b; q2 += c * c; q3 += d * d;
  }
  float* ps = part + (size_t)blockIdx.x * 2048;
  ps[t] = s0; ps[t + 256] = s1; ps[t + 512] = s2; ps[t + 768] = s3;
  ps[1024 + t] = q0; ps[1024 + t + 256] = q1; ps[1024 + t + 512] = q2; ps[1024 + t + 768] = q3;
}

__global__ void k_colreduce(const float* __restrict__ part, float* __restrict__ sums, float* __restrict__ sumsq)
{
  int c = blockIdx.x * blockDim.x + threadIdx.x;   // 0..1023
  float s = 0, q = 0;
  for (int i = 0; i < 16; ++i) { s += part[(size_t)i * 2048 + c]; q += part[(size_t)i * 2048 + 1024 + c]; }
  sums[c] = s; sumsq[c] = q;
}

__global__ void k_bn_quant(const float* __restrict__ y, const float* __restrict__ sums,
                           const float* __restrict__ sumsq, const float* __restrict__ g,
                           const float* __restrict__ bb, const float* __restrict__ aq5,
                           float* __restrict__ out1, __bf16* __restrict__ outbf, int n)
{
  int i = blockIdx.x * blockDim.x + threadIdx.x;
  if (i >= n) return;
  int ch = i & 1023;
  float mean = sums[ch] * (1.f / 4096.f);
  float var = sumsq[ch] * (1.f / 4096.f) - mean * mean;
  float v = (y[i] - mean) * rsqrtf(var + 1e-6f) * g[ch] + bb[ch];
  v = lsq_fwd(v, aq5[0]);
  out1[i] = v;
  outbf[i] = (__bf16)v;
}

// ---------------- host ----------------
extern "C" void kernel_launch(void* const* d_in, const int* in_sizes, int n_in,
                              void* d_out, int out_size, void* d_ws, size_t ws_size,
                              hipStream_t stream)
{
  const float* q    = (const float*)d_in[0];
  const float* k    = (const float*)d_in[1];
  const float* v    = (const float*)d_in[2];
  const int*   mask = (const int*)  d_in[3];
  const float* wq_w = (const float*)d_in[4];
  const float* wk_w = (const float*)d_in[5];
  const float* wv_w = (const float*)d_in[6];
  const float* fc_w = (const float*)d_in[7];
  const float* wq_a = (const float*)d_in[8];
  const float* wk_a = (const float*)d_in[9];
  const float* wv_a = (const float*)d_in[10];
  const float* fc_a = (const float*)d_in[11];
  const float* aq1  = (const float*)d_in[12];
  const float* aq2  = (const float*)d_in[13];
  const float* aq3  = (const float*)d_in[14];
  const float* aq4  = (const float*)d_in[15];
  const float* aq5  = (const float*)d_in[16];
  const float* bn_g = (const float*)d_in[17];
  const float* bn_b = (const float*)d_in[18];
  const float* fa1_w = (const float*)d_in[19];
  const float* fa1_b = (const float*)d_in[20];
  const float* fa2_w = (const float*)d_in[21];
  const float* fa2_b = (const float*)d_in[22];
  const float* fa3_w = (const float*)d_in[23];
  const float* fa3_b = (const float*)d_in[24];

  char* ws = (char*)d_ws;
  __bf16* WQB   = (__bf16*)(ws + 0);
  __bf16* WKB   = (__bf16*)(ws + 2097152);
  __bf16* WVB   = (__bf16*)(ws + 4194304);
  __bf16* FCB   = (__bf16*)(ws + 6291456);
  __bf16* QBF   = (__bf16*)(ws + 8388608);
  __bf16* KBF   = (__bf16*)(ws + 16777216);
  __bf16* VBF   = (__bf16*)(ws + 25165824);
  __bf16* QPb   = (__bf16*)(ws + 33554432);
  __bf16* KPb   = (__bf16*)(ws + 41943040);
  __bf16* VPT   = (__bf16*)(ws + 50331648);
  float*  ROWMX = (float*) (ws + 58720256);
  float*  ROWSM = (float*) (ws + 58982400);
  float*  PART  = (float*) (ws + 59244544);
  float*  SUMS  = (float*) (ws + 59375616);
  float*  SUMSQ = (float*) (ws + 59379712);
  __bf16* FA1WB = (__bf16*)(ws + 59383808);
  __bf16* FA2WB = (__bf16*)(ws + 59514880);
  __bf16* FA3WB = (__bf16*)(ws + 59523072);
  // aliases (lifetimes disjoint)
  __bf16* AO    = QBF;                         // attn@V output, after q-proj consumed QBF
  float*  Y     = (float*)(ws + 16777216);     // fc+resid fp32, overlays KBF+VBF
  __bf16* OUTBF = QPb;                         // bf16 copy of out1, after attention done
  __bf16* FA1O  = KPb;                         // 4096x64
  __bf16* FA2O  = (__bf16*)(ws + 41943040 + 524288);

  float* OUT1 = (float*)d_out;
  float* ATTN = (float*)d_out + 4194304;
  float* FAO  = (float*)d_out + 71303168;

  // 1) weight quantization + bf16 conversions
  k_quant_w<<<4096, 256, 0, stream>>>(wq_w, wq_a, WQB, 1048576);
  k_quant_w<<<4096, 256, 0, stream>>>(wk_w, wk_a, WKB, 1048576);
  k_quant_w<<<4096, 256, 0, stream>>>(wv_w, wv_a, WVB, 1048576);
  k_quant_w<<<4096, 256, 0, stream>>>(fc_w, fc_a, FCB, 1048576);
  k_cvt_bf16<<<16384, 256, 0, stream>>>(q, QBF, 4194304);
  k_cvt_bf16<<<16384, 256, 0, stream>>>(k, KBF, 4194304);
  k_cvt_bf16<<<16384, 256, 0, stream>>>(v, VBF, 4194304);
  k_cvt_bf16<<<256, 256, 0, stream>>>(fa1_w, FA1WB, 65536);
  k_cvt_bf16<<<16, 256, 0, stream>>>(fa2_w, FA2WB, 4096);
  k_cvt_bf16<<<256, 256, 0, stream>>>(fa3_w, FA3WB, 65536);

  // 2) projections (quantized outputs); V goes out per-head transposed
  k_gemm_xwt<<<512, 256, 0, stream>>>(QBF, WQB, 4096, 1024, 1024, 0, aq1, nullptr, nullptr, nullptr, QPb, nullptr);
  k_gemm_xwt<<<512, 256, 0, stream>>>(KBF, WKB, 4096, 1024, 1024, 0, aq2, nullptr, nullptr, nullptr, KPb, nullptr);
  k_gemm_xwt<<<512, 256, 0, stream>>>(VBF, WVB, 4096, 1024, 1024, 1, aq3, nullptr, nullptr, nullptr, nullptr, VPT);

  // 3) attention: pass 1 stats, pass 2 write-P-once + P@V
  k_attn_stats<<<512, 256, 0, stream>>>(QPb, KPb, mask, ROWMX, ROWSM);
  k_attn_pv<<<512, 256, 0, stream>>>(QPb, KPb, VPT, mask, ROWMX, ROWSM, ATTN, AO);

  // 4) fc projection + quant + residual
  k_gemm_xwt<<<512, 256, 0, stream>>>(AO, FCB, 4096, 1024, 1024, 2, aq4, nullptr, q, Y, nullptr, nullptr);

  // 5) batchnorm stats (deterministic two-stage) + normalize + quant
  k_colstats<<<16, 256, 0, stream>>>(Y, PART);
  k_colreduce<<<4, 256, 0, stream>>>(PART, SUMS, SUMSQ);
  k_bn_quant<<<16384, 256, 0, stream>>>(Y, SUMS, SUMSQ, bn_g, bn_b, aq5, OUT1, OUTBF, 4194304);

  // 6) low-rank FFN chain + skip
  k_gemm_xwt<<<32, 256, 0, stream>>>(OUTBF, FA1WB, 4096, 64, 1024, 3, nullptr, fa1_b, nullptr, nullptr, FA1O, nullptr);
  k_gemm_xwt<<<32, 256, 0, stream>>>(FA1O, FA2WB, 4096, 64, 64, 3, nullptr, fa2_b, nullptr, nullptr, FA2O, nullptr);
  k_gemm_xwt<<<512, 256, 0, stream>>>(FA2O, FA3WB, 4096, 1024, 64, 4, nullptr, fa3_b, OUT1, FAO, nullptr, nullptr);
}
